// MultiHeadDifferentialAttention_13417477833076
// MI455X (gfx1250) — matrix operand order repaired and checked by reading, not compiled, not run on hardware
//
#include <hip/hip_runtime.h>

// ---------------------------------------------------------------------------
// MultiHeadDifferentialAttention for MI455X (gfx1250, wave32, WMMA)
// B=1, S=2048, E=1024, H=16, D=64
// All heavy math on v_wmma_f32_16x16x32_bf16. B-side operands are kept in a
// fragment-major ("swizzled") layout so every fragment load is 2x b128.
// GEMM waves compute 64x64 strips: 16 WMMAs per K-step, ~1 vmem per WMMA.
// ---------------------------------------------------------------------------

#define SQ 2048
#define EE 1024
#define HH 16
#define DD 64

typedef __bf16        v16bf __attribute__((ext_vector_type(16)));
typedef float         v8f   __attribute__((ext_vector_type(8)));
typedef unsigned int  v4u   __attribute__((ext_vector_type(4)));

union AFrag { v4u q[2]; unsigned int u[8]; v16bf v; };
union BFrag { v4u q[2]; unsigned int u[8]; v16bf v; };
union CFrag { v8f v; float f[8]; };

#define WMMA_BF16(A, B, C) \
  __builtin_amdgcn_wmma_f32_16x16x32_bf16(false, (A), false, (B), (short)0, (C), false, false)

__device__ __forceinline__ unsigned short f2bf(float f) {
  unsigned int u = __float_as_uint(f);
  u += 0x7FFFu + ((u >> 16) & 1u);   // round-to-nearest-even
  return (unsigned short)(u >> 16);
}

// A-matrix 16x32 bf16 fragment (ISA 7.12.2), row-major source, 16B aligned:
// lanes 0-15 -> M=lane; VGPR0-7 = K 0..15 (+16 upper half): contiguous order of this code, not the ISA table order.
__device__ __forceinline__ void load_a_frag(AFrag& a, const unsigned short* base, int ld) {
  const int lane = threadIdx.x & 31;
  const int m = lane & 15, kh = lane >> 4;
  const unsigned short* p = base + (size_t)m * ld + kh * 16;
  a.q[0] = *reinterpret_cast<const v4u*>(p);
  a.q[1] = *reinterpret_cast<const v4u*>(p + 8);
}

// Swizzled B layout: for a KxN matrix tiled into 32x16 B tiles,
//   tile  = (n/16)*(K/32) + (k/32)
//   lane  = (n&15) + 16*((k%32)/16)      (B layout: N = lane&15, K-half = lane>>4)
//   off   = k % 16                        (pairs packed per VGPR)
// element stored at  [tile*32 + lane]*16 + off  -> one 32B chunk per lane.
__device__ __forceinline__ void load_b_frag_swz(BFrag& b, const unsigned short* mat,
                                                int K, int tileK, int tileN) {
  const int lane = threadIdx.x & 31;
  const unsigned short* p = mat + (((size_t)tileN * (K >> 5) + tileK) * 32 + lane) * 16;
  b.q[0] = *reinterpret_cast<const v4u*>(p);
  b.q[1] = *reinterpret_cast<const v4u*>(p + 8);
}

// ---------------------------------------------------------------------------
// f32 -> bf16 row-major conversion (A-side operands)
// ---------------------------------------------------------------------------
__global__ void cvt_bf16_kernel(const float* __restrict__ src,
                                unsigned short* __restrict__ dst, int n) {
  for (int i = blockIdx.x * blockDim.x + threadIdx.x; i < n; i += gridDim.x * blockDim.x)
    dst[i] = f2bf(src[i]);
}

// ---------------------------------------------------------------------------
// f32 -> bf16 convert + swizzle into fragment-major B layout.
// src: batch x K x N row-major. dst: batch x (K*N) swizzled.
// ---------------------------------------------------------------------------
__global__ void swz_b_kernel(const float* __restrict__ src, unsigned short* __restrict__ dst,
                             int K, int N, int batch) {
  const long total = (long)batch * K * N;
  const long stride = (long)gridDim.x * blockDim.x;
  for (long i = blockIdx.x * blockDim.x + threadIdx.x; i < total; i += stride) {
    const int  b   = (int)(i / ((long)K * N));
    const int  rem = (int)(i % ((long)K * N));
    const int  e = rem / N, c = rem % N;
    const int  tile = (c >> 4) * (K >> 5) + (e >> 5);
    const int  kl = e & 31;
    const int  lane = (c & 15) + ((kl >> 4) << 4);
    const int  off = kl & 15;
    dst[(size_t)b * K * N + ((size_t)tile * 32 + lane) * 16 + off] = f2bf(src[i]);
  }
}

// ---------------------------------------------------------------------------
// QKV projection. Each wave computes a 64x64 strip: 4 A fragments (M-tiles)
// re-use each B fragment, 16 WMMAs per K-step. Column groups of 64 never
// straddle Q/K/V sections (320 = 5*64).
// Q  -> row-major [h][s][128] bf16            (A-side of the score WMMAs)
// K  -> swizzled B layout per head, K=128 (d rows), N=2048 (kpos)
// V  -> swizzled B layout per head, K=2048 (kpos rows), N=64 (d)
// ---------------------------------------------------------------------------
__global__ __launch_bounds__(32) void qkv_gemm_kernel(
    const unsigned short* __restrict__ Xb,
    const unsigned short* __restrict__ Wqs, const unsigned short* __restrict__ Wks,
    const unsigned short* __restrict__ Wvs,
    const float* __restrict__ bq, const float* __restrict__ bk, const float* __restrict__ bv,
    unsigned short* __restrict__ Qb, unsigned short* __restrict__ Ksw,
    unsigned short* __restrict__ Vsw) {
  const int g0 = blockIdx.x * 64;       // 0..5119, group of 4 column tiles
  const int s0 = blockIdx.y * 64;       // 64-row strip
  const int head = g0 / 320;
  const int w = g0 % 320;               // 0,64 -> Q ; 128,192 -> K ; 256 -> V

  const unsigned short* Wsec; const float* bias; int mode, colbase;
  if (w < 128)      { Wsec = Wqs + (size_t)head * EE * 128; bias = bq + head * 128; mode = 0; colbase = w; }
  else if (w < 256) { Wsec = Wks + (size_t)head * EE * 128; bias = bk + head * 128; mode = 1; colbase = w - 128; }
  else              { Wsec = Wvs + (size_t)head * EE * 64;  bias = bv + head * 64;  mode = 2; colbase = 0; }

  CFrag acc[4][4];                      // [mtile][ntile]
#pragma unroll
  for (int mt = 0; mt < 4; ++mt)
#pragma unroll
    for (int t = 0; t < 4; ++t)
#pragma unroll
      for (int r = 0; r < 8; ++r) acc[mt][t].f[r] = 0.f;

#pragma unroll 1
  for (int e = 0; e < EE; e += 32) {
    __builtin_prefetch(Wsec + (((size_t)(colbase >> 4) * (EE >> 5) + (e >> 5) + 1) * 32) * 16, 0, 1);
    AFrag a[4];
#pragma unroll
    for (int mt = 0; mt < 4; ++mt)
      load_a_frag(a[mt], Xb + (size_t)(s0 + mt * 16) * EE + e, EE);
#pragma unroll
    for (int t = 0; t < 4; ++t) {
      BFrag b; load_b_frag_swz(b, Wsec, EE, e >> 5, (colbase >> 4) + t);
#pragma unroll
      for (int mt = 0; mt < 4; ++mt)
        acc[mt][t].v = WMMA_BF16(a[mt].v, b.v, acc[mt][t].v);
    }
  }

  const int lane = threadIdx.x & 31;
  const int n = lane & 15, mhalf = lane >> 4;
#pragma unroll
  for (int mt = 0; mt < 4; ++mt)
#pragma unroll
    for (int t = 0; t < 4; ++t) {
      const int col = colbase + t * 16;
      const float bval = bias[col + n];
#pragma unroll
      for (int r = 0; r < 8; ++r) {
        const int m = mt * 16 + r + 8 * mhalf;      // row within 64-strip
        const unsigned short o = f2bf(acc[mt][t].f[r] + bval);
        if (mode == 0) {
          Qb[((size_t)head * SQ + s0 + m) * 128 + col + n] = o;
        } else if (mode == 1) {
          // K^T swizzled: per-head B matrix, K=128 (d=col+n), N=2048 (kpos)
          const int d = col + n, kpos = s0 + m;
          const int tile = (kpos >> 4) * 4 + (d >> 5);
          const int kl = d & 31;
          const int ldst = (kpos & 15) + ((kl >> 4) << 4);
          Ksw[(size_t)head * 128 * SQ + ((size_t)tile * 32 + ldst) * 16 + (kl & 15)] = o;
        } else {
          // V swizzled: per-head B matrix, K=2048 (kpos), N=64 (d=col+n)
          const int d = col + n, kpos = s0 + m;
          const int tile = (d >> 4) * (SQ >> 5) + (kpos >> 5);
          const int kl = kpos & 31;
          const int ldst = (d & 15) + ((kl >> 4) << 4);
          Vsw[(size_t)head * SQ * DD + ((size_t)tile * 32 + ldst) * 16 + (kl & 15)] = o;
        }
      }
    }
}

// ---------------------------------------------------------------------------
// Flash-style differential attention. One wave owns a 16-row query tile of
// one head. Two independent online softmaxes; final O = O1/l1 - lam*O2/l2.
// All fragment loads are b128; P re-layouts C->A through a 1KB LDS tile.
// ---------------------------------------------------------------------------
__device__ __forceinline__ void branch_update(
    CFrag& lo, CFrag& hi, float (&rm)[8], float (&rl)[8], CFrag (&oacc)[4],
    const BFrag (&vf)[4], unsigned short* Plds) {
  const int lane = threadIdx.x & 31;
  const int n = lane & 15, mhalf = lane >> 4;
#pragma unroll
  for (int r = 0; r < 8; ++r) {
    float a = lo.f[r] * 0.125f;                     // 1/sqrt(64)
    float b = hi.f[r] * 0.125f;
    float mx = fmaxf(a, b);                         // row-max across 16 lanes
    mx = fmaxf(mx, __shfl_xor(mx, 1));
    mx = fmaxf(mx, __shfl_xor(mx, 2));
    mx = fmaxf(mx, __shfl_xor(mx, 4));
    mx = fmaxf(mx, __shfl_xor(mx, 8));
    float nm = fmaxf(rm[r], mx);
    float corr = __expf(rm[r] - nm);
    rm[r] = nm;
    float pa = __expf(a - nm);
    float pb = __expf(b - nm);
    float rs = pa + pb;
    rs += __shfl_xor(rs, 1);
    rs += __shfl_xor(rs, 2);
    rs += __shfl_xor(rs, 4);
    rs += __shfl_xor(rs, 8);
    rl[r] = rl[r] * corr + rs;
#pragma unroll
    for (int t = 0; t < 4; ++t) oacc[t].f[r] *= corr;
    const int m = r + 8 * mhalf;
    Plds[m * 32 + n]      = f2bf(pa);               // kpos j..j+15
    Plds[m * 32 + 16 + n] = f2bf(pb);               // kpos j+16..j+31
  }
  // single wave: DS ops are in-order, loads below see the stores above
  AFrag p;
  load_a_frag(p, Plds, 32);
#pragma unroll
  for (int t = 0; t < 4; ++t)
    oacc[t].v = WMMA_BF16(p.v, vf[t].v, oacc[t].v);
}

__global__ __launch_bounds__(32) void diffattn_kernel(
    const unsigned short* __restrict__ Qb, const unsigned short* __restrict__ Ksw,
    const unsigned short* __restrict__ Vsw, const float* __restrict__ lam_p,
    unsigned short* __restrict__ Ocat) {
  __shared__ __align__(16) unsigned short Plds[16 * 32];
  const int h = blockIdx.y;
  const int s0 = blockIdx.x * 16;
  const float lam = lam_p[0];

  const unsigned short* Qbase = Qb + ((size_t)h * SQ + s0) * 128;
  AFrag q1a, q1b, q2a, q2b;                         // Q1 = cols 0..63, Q2 = 64..127
  load_a_frag(q1a, Qbase + 0, 128);
  load_a_frag(q1b, Qbase + 32, 128);
  load_a_frag(q2a, Qbase + 64, 128);
  load_a_frag(q2b, Qbase + 96, 128);

  CFrag o1[4], o2[4];
  float m1[8], l1[8], m2[8], l2[8];
#pragma unroll
  for (int r = 0; r < 8; ++r) { m1[r] = -1e30f; l1[r] = 0.f; m2[r] = -1e30f; l2[r] = 0.f; }
#pragma unroll
  for (int t = 0; t < 4; ++t)
#pragma unroll
    for (int r = 0; r < 8; ++r) { o1[t].f[r] = 0.f; o2[t].f[r] = 0.f; }

  const unsigned short* Kh = Ksw + (size_t)h * 128 * SQ;  // dt 0..1 = K1, 2..3 = K2
  const unsigned short* Vh = Vsw + (size_t)h * SQ * DD;

#pragma unroll 1
  for (int j = 0; j < SQ; j += 32) {
    const int st = j >> 4;
    CFrag sAlo, sAhi, sBlo, sBhi;
#pragma unroll
    for (int r = 0; r < 8; ++r) { sAlo.f[r] = 0.f; sAhi.f[r] = 0.f; sBlo.f[r] = 0.f; sBhi.f[r] = 0.f; }

    BFrag bf;
    // branch 1: Q1 x K1^T (d tiles 0,1), kpos tiles st (lo) and st+1 (hi)
    load_b_frag_swz(bf, Kh, 128, 0, st);     sAlo.v = WMMA_BF16(q1a.v, bf.v, sAlo.v);
    load_b_frag_swz(bf, Kh, 128, 1, st);     sAlo.v = WMMA_BF16(q1b.v, bf.v, sAlo.v);
    load_b_frag_swz(bf, Kh, 128, 0, st + 1); sAhi.v = WMMA_BF16(q1a.v, bf.v, sAhi.v);
    load_b_frag_swz(bf, Kh, 128, 1, st + 1); sAhi.v = WMMA_BF16(q1b.v, bf.v, sAhi.v);
    // branch 2: Q2 x K2^T (d tiles 2,3)
    load_b_frag_swz(bf, Kh, 128, 2, st);     sBlo.v = WMMA_BF16(q2a.v, bf.v, sBlo.v);
    load_b_frag_swz(bf, Kh, 128, 3, st);     sBlo.v = WMMA_BF16(q2b.v, bf.v, sBlo.v);
    load_b_frag_swz(bf, Kh, 128, 2, st + 1); sBhi.v = WMMA_BF16(q2a.v, bf.v, sBhi.v);
    load_b_frag_swz(bf, Kh, 128, 3, st + 1); sBhi.v = WMMA_BF16(q2b.v, bf.v, sBhi.v);

    // V fragments (rows j..j+31, d tiles 0..3) shared by both branches
    BFrag vf[4];
#pragma unroll
    for (int t = 0; t < 4; ++t) load_b_frag_swz(vf[t], Vh, SQ, j >> 5, t);

    branch_update(sAlo, sAhi, m1, l1, o1, vf, Plds);
    branch_update(sBlo, sBhi, m2, l2, o2, vf, Plds);
  }

  const int lane = threadIdx.x & 31;
  const int n = lane & 15, mhalf = lane >> 4;
#pragma unroll
  for (int t = 0; t < 4; ++t)
#pragma unroll
    for (int r = 0; r < 8; ++r) {
      const int m = r + 8 * mhalf;
      float val = o1[t].f[r] / l1[r] - lam * (o2[t].f[r] / l2[r]);
      Ocat[(size_t)(s0 + m) * EE + h * DD + t * 16 + n] = f2bf(val);
    }
}

// ---------------------------------------------------------------------------
// Output projection: Y[s,c] = sum_k Ocat[s,k] * Wo_swz[k,c] + bo[c] (f32 out).
// Each wave computes a 64x64 strip (16 WMMAs per K-step).
// ---------------------------------------------------------------------------
__global__ __launch_bounds__(32) void oproj_kernel(
    const unsigned short* __restrict__ Ocat, const unsigned short* __restrict__ Wos,
    const float* __restrict__ bo, float* __restrict__ Y) {
  const int c0 = blockIdx.x * 64;
  const int s0 = blockIdx.y * 64;
  CFrag acc[4][4];
#pragma unroll
  for (int mt = 0; mt < 4; ++mt)
#pragma unroll
    for (int t = 0; t < 4; ++t)
#pragma unroll
      for (int r = 0; r < 8; ++r) acc[mt][t].f[r] = 0.f;

#pragma unroll 1
  for (int e = 0; e < EE; e += 32) {
    __builtin_prefetch(Wos + (((size_t)(c0 >> 4) * (EE >> 5) + (e >> 5) + 1) * 32) * 16, 0, 1);
    AFrag a[4];
#pragma unroll
    for (int mt = 0; mt < 4; ++mt)
      load_a_frag(a[mt], Ocat + (size_t)(s0 + mt * 16) * EE + e, EE);
#pragma unroll
    for (int t = 0; t < 4; ++t) {
      BFrag b; load_b_frag_swz(b, Wos, EE, e >> 5, (c0 >> 4) + t);
#pragma unroll
      for (int mt = 0; mt < 4; ++mt)
        acc[mt][t].v = WMMA_BF16(a[mt].v, b.v, acc[mt][t].v);
    }
  }
  const int lane = threadIdx.x & 31;
  const int n = lane & 15, mhalf = lane >> 4;
#pragma unroll
  for (int mt = 0; mt < 4; ++mt)
#pragma unroll
    for (int t = 0; t < 4; ++t) {
      const float bias = bo[c0 + t * 16 + n];
#pragma unroll
      for (int r = 0; r < 8; ++r) {
        const int m = mt * 16 + r + 8 * mhalf;
        Y[(size_t)(s0 + m) * EE + c0 + t * 16 + n] = acc[mt][t].f[r] + bias;
      }
    }
}

// ---------------------------------------------------------------------------
// LayerNorm over E=1024 per row, times (1 - LAM_INIT) = 0.2
// ---------------------------------------------------------------------------
__global__ __launch_bounds__(256) void ln_kernel(
    const float* __restrict__ Y, const float* __restrict__ gamma,
    const float* __restrict__ beta, float* __restrict__ out) {
  __shared__ float s_sum[256], s_sq[256];
  const int row = blockIdx.x;
  const float* y = Y + (size_t)row * EE;
  float sum = 0.f, sq = 0.f;
  for (int i = threadIdx.x; i < EE; i += 256) {
    float v = y[i];
    sum += v; sq += v * v;
  }
  s_sum[threadIdx.x] = sum; s_sq[threadIdx.x] = sq;
  __syncthreads();
  for (int o = 128; o > 0; o >>= 1) {
    if ((int)threadIdx.x < o) {
      s_sum[threadIdx.x] += s_sum[threadIdx.x + o];
      s_sq[threadIdx.x]  += s_sq[threadIdx.x + o];
    }
    __syncthreads();
  }
  const float mu  = s_sum[0] * (1.f / EE);
  const float var = s_sq[0] * (1.f / EE) - mu * mu;
  const float inv = rsqrtf(var + 1e-5f);
  for (int i = threadIdx.x; i < EE; i += 256)
    out[(size_t)row * EE + i] = ((y[i] - mu) * inv * gamma[i] + beta[i]) * 0.2f;
}

// ---------------------------------------------------------------------------
extern "C" void kernel_launch(void* const* d_in, const int* in_sizes, int n_in,
                              void* d_out, int out_size, void* d_ws, size_t ws_size,
                              hipStream_t stream) {
  (void)in_sizes; (void)n_in; (void)out_size; (void)ws_size;
  const float* X     = (const float*)d_in[0];
  const float* Wq    = (const float*)d_in[1];
  const float* bq    = (const float*)d_in[2];
  const float* Wk    = (const float*)d_in[3];
  const float* bk    = (const float*)d_in[4];
  const float* Wv    = (const float*)d_in[5];
  const float* bv    = (const float*)d_in[6];
  const float* Wo    = (const float*)d_in[7];
  const float* bo    = (const float*)d_in[8];
  const float* gamma = (const float*)d_in[9];
  const float* beta  = (const float*)d_in[10];
  const float* lam   = (const float*)d_in[11];

  char* ws = (char*)d_ws;
  size_t off = 0;
  auto take = [&](size_t bytes) -> char* {
    char* p = ws + off;
    off = (off + bytes + 255) & ~(size_t)255;
    return p;
  };
  unsigned short* Xb  = (unsigned short*)take((size_t)SQ * EE * 2);        // row-major A
  unsigned short* Wqs = (unsigned short*)take((size_t)HH * EE * 128 * 2);  // swizzled B
  unsigned short* Wks = (unsigned short*)take((size_t)HH * EE * 128 * 2);  // swizzled B
  unsigned short* Wvs = (unsigned short*)take((size_t)HH * EE * 64 * 2);   // swizzled B
  unsigned short* Wos = (unsigned short*)take((size_t)EE * EE * 2);        // swizzled B
  unsigned short* Qb  = (unsigned short*)take((size_t)HH * SQ * 128 * 2);  // row-major A
  unsigned short* Ksw = (unsigned short*)take((size_t)HH * SQ * 128 * 2);  // swizzled B
  unsigned short* Vsw = (unsigned short*)take((size_t)HH * SQ * 64 * 2);   // swizzled B
  unsigned short* Oc  = (unsigned short*)take((size_t)SQ * EE * 2);        // row-major A
  float*          Y   = (float*)take((size_t)SQ * EE * 4);

  cvt_bf16_kernel<<<1024, 256, 0, stream>>>(X, Xb, SQ * EE);
  swz_b_kernel<<<1024, 256, 0, stream>>>(Wq, Wqs, EE, 128, HH);
  swz_b_kernel<<<1024, 256, 0, stream>>>(Wk, Wks, EE, 128, HH);
  swz_b_kernel<<<512,  256, 0, stream>>>(Wv, Wvs, EE, 64, HH);
  swz_b_kernel<<<512,  256, 0, stream>>>(Wo, Wos, EE, EE, 1);

  qkv_gemm_kernel<<<dim3(80, 32), 32, 0, stream>>>(Xb, Wqs, Wks, Wvs,
                                                   bq, bk, bv, Qb, Ksw, Vsw);
  diffattn_kernel<<<dim3(128, 16), 32, 0, stream>>>(Qb, Ksw, Vsw, lam, Oc);
  oproj_kernel<<<dim3(16, 32), 32, 0, stream>>>(Oc, Wos, bo, Y);
  ln_kernel<<<SQ, 256, 0, stream>>>(Y, gamma, beta, (float*)d_out);
}
